// MLA_85280870629747
// MI455X (gfx1250) — matrix operand order repaired and checked by reading, not compiled, not run on hardware
//
#include <hip/hip_runtime.h>
#include <hip/hip_bf16.h>
#include <cmath>

// ---------------------------------------------------------------------------
// Problem constants (match reference)
// ---------------------------------------------------------------------------
constexpr int cB  = 4;
constexpr int cT  = 2048;   // power of 2 (epilogue uses shift/mask)
constexpr int cV  = 2048;
constexpr int cD  = 512;
constexpr int cH  = 8;
constexpr int cL  = 4;
constexpr int cHD = 64;

typedef __attribute__((ext_vector_type(16))) __bf16 v16bf;
typedef __attribute__((ext_vector_type(8)))  __bf16 v8bf;
typedef __attribute__((ext_vector_type(8)))  float  v8f;

__device__ __forceinline__ v8f v8f_zero() {
  v8f z;
#pragma unroll
  for (int i = 0; i < 8; ++i) z[i] = 0.0f;
  return z;
}

__device__ __forceinline__ v8f wmma_bf16(v16bf a, v16bf b, v8f c) {
  // v_wmma_f32_16x16x32_bf16 : D = A(16x32) * B(32x16) + C
  return __builtin_amdgcn_wmma_f32_16x16x32_bf16(
      /*neg_a=*/false, a, /*neg_b=*/false, b,
      /*c_mod=*/(short)0, c, /*reuse_a=*/false, /*reuse_b=*/false);
}

// ds_swizzle_b32 XOR butterfly (BITMASK mode: and=0x1f, or=0, xor=XM).
// Masks 1/2/4/8 never cross the 16-lane half-group boundary.
template <int XM>
__device__ __forceinline__ float swz_xor(float v) {
  int i = __builtin_bit_cast(int, v);
  i = __builtin_amdgcn_ds_swizzle(i, (XM << 10) | 0x1F);
  return __builtin_bit_cast(float, i);
}

__device__ __forceinline__ float grp16_max(float v) {
  v = fmaxf(v, swz_xor<8>(v));
  v = fmaxf(v, swz_xor<4>(v));
  v = fmaxf(v, swz_xor<2>(v));
  v = fmaxf(v, swz_xor<1>(v));
  return v;
}

__device__ __forceinline__ float grp16_sum(float v) {
  v += swz_xor<8>(v);
  v += swz_xor<4>(v);
  v += swz_xor<2>(v);
  v += swz_xor<1>(v);
  return v;
}

// ---------------------------------------------------------------------------
// Fragment loaders (this code's contiguous k order, same for A and B; wave32)
//
// A (16x32, bf16): lane holds row m = lane&15.
//   lanes 0-15 : elems i -> K=i ; lanes 16-31: elems i -> K=16+i
// B (32x16, bf16): lane holds col n = lane&15.
//   lanes 0-15 : elems i -> K=i ; lanes 16-31: elems i -> K=16+i
// ---------------------------------------------------------------------------
__device__ __forceinline__ v16bf load_a_bf16(const __bf16* A, int lda,
                                             int mbase, int kbase, int lane) {
  const int m    = mbase + (lane & 15);
  const int koff = kbase + ((lane >> 4) << 4);      // +16 for upper half
  const __bf16* p = A + (size_t)m * lda + koff;
  v8bf lo = *(const v8bf*)p;                        // K = koff .. koff+7
  v8bf hi = *(const v8bf*)(p + 8);                  // K = koff+8 .. koff+15
  v16bf r;
#pragma unroll
  for (int i = 0; i < 8; ++i) { r[i] = lo[i]; r[i + 8] = hi[i]; }
  return r;
}

__device__ __forceinline__ v16bf load_a_f32(const float* A, int lda,
                                            int mbase, int kbase, int lane) {
  const int m    = mbase + (lane & 15);
  const int koff = kbase + ((lane >> 4) << 4);
  const float* p = A + (size_t)m * lda + koff;
  v16bf r;
#pragma unroll
  for (int i = 0; i < 8; ++i) {
    r[i]     = (__bf16)p[i];
    r[i + 8] = (__bf16)p[i + 8];
  }
  return r;
}

// B[k,n] = W[nbase+n][kbase+k] : 16 contiguous bf16 per lane (32 bytes)
__device__ __forceinline__ v16bf load_b_rowK(const __bf16* W, int ldw,
                                             int nbase, int kbase, int lane) {
  const int n = nbase + (lane & 15);
  const int k = kbase + ((lane >> 4) << 4);         // +16 for upper half
  return *(const v16bf*)(W + (size_t)n * ldw + k);
}

// ---------------------------------------------------------------------------
// f32 -> bf16 conversion
// ---------------------------------------------------------------------------
__global__ void cvt_f32_bf16(const float* __restrict__ in,
                             __bf16* __restrict__ out, size_t n) {
  size_t i = (size_t)blockIdx.x * blockDim.x + threadIdx.x;
  if (i < n) out[i] = (__bf16)in[i];
}

// ---------------------------------------------------------------------------
// Generic WMMA GEMM: C[m,n] = scale * sum_k A[m,k] * W[n,k]   (= A @ W^T)
// 256 threads = 8 waves as 2(m) x 4(n); wave tile 32x64 (2x4 WMMA tiles).
// WG tile: 64(m) x 256(n). grid = (N/256, M/64).
// MODE 0: out[m*N+n] (+pos[t,n] if POS)           row-major [M,N]
// MODE 1: out[((b*H+h)*T+t)*HD+hd]                head-split (Q/K)
// MODE 2: out[((b*H+h)*HD+hd)*T+t]                head-split transposed (V)
// ---------------------------------------------------------------------------
template <bool AF32, int MODE, bool POS>
__global__ __launch_bounds__(256) void wmma_gemm(
    const void* __restrict__ Ap, const __bf16* __restrict__ W,
    __bf16* __restrict__ out, int M, int N, int K, float scale,
    const float* __restrict__ pos) {
  const int lane = threadIdx.x & 31;
  const int wid  = threadIdx.x >> 5;
  const int mbase = blockIdx.y * 64  + (wid & 1) * 32;
  const int nbase = blockIdx.x * 256 + (wid >> 1) * 64;

  v8f acc[2][4];
#pragma unroll
  for (int mi = 0; mi < 2; ++mi)
#pragma unroll
    for (int ni = 0; ni < 4; ++ni) acc[mi][ni] = v8f_zero();

  const float*  Af = (const float*)Ap;
  const __bf16* Ab = (const __bf16*)Ap;

#pragma unroll 2
  for (int k0 = 0; k0 < K; k0 += 32) {
    if (k0 + 32 < K) {  // global_prefetch_b8 of next W tile
      __builtin_prefetch(W + (size_t)(nbase + (lane & 15)) * K + k0 + 32, 0, 0);
    }
    v16bf a[2];
    if constexpr (AF32) {
      a[0] = load_a_f32(Af, K, mbase,      k0, lane);
      a[1] = load_a_f32(Af, K, mbase + 16, k0, lane);
    } else {
      a[0] = load_a_bf16(Ab, K, mbase,      k0, lane);
      a[1] = load_a_bf16(Ab, K, mbase + 16, k0, lane);
    }
    v16bf b[4];
#pragma unroll
    for (int ni = 0; ni < 4; ++ni)
      b[ni] = load_b_rowK(W, K, nbase + ni * 16, k0, lane);
#pragma unroll
    for (int mi = 0; mi < 2; ++mi)
#pragma unroll
      for (int ni = 0; ni < 4; ++ni)
        acc[mi][ni] = wmma_bf16(a[mi], b[ni], acc[mi][ni]);
  }

  // D layout: vgpr r, lanes 0-15 -> (m=r, n=lane) ; lanes 16-31 -> (m=r+8)
  const int mofs = (lane >> 4) << 3;
  const int ncol = lane & 15;
#pragma unroll
  for (int mi = 0; mi < 2; ++mi) {
#pragma unroll
    for (int ni = 0; ni < 4; ++ni) {
#pragma unroll
      for (int r = 0; r < 8; ++r) {
        const int m = mbase + mi * 16 + r + mofs;
        const int n = nbase + ni * 16 + ncol;
        float val = acc[mi][ni][r] * scale;
        if constexpr (MODE == 0) {
          if constexpr (POS) val += pos[(size_t)(m & (cT - 1)) * N + n];
          out[(size_t)m * N + n] = (__bf16)val;
        } else if constexpr (MODE == 1) {
          const int b = m >> 11, t = m & (cT - 1);     // cT = 2048
          const int hh = n >> 6, hd = n & 63;          // cHD = 64
          out[(((size_t)b * cH + hh) * cT + t) * cHD + hd] = (__bf16)val;
        } else {
          const int b = m >> 11, t = m & (cT - 1);
          const int hh = n >> 6, hd = n & 63;
          out[(((size_t)b * cH + hh) * cHD + hd) * (size_t)cT + t] = (__bf16)val;
        }
      }
    }
  }
}

// ---------------------------------------------------------------------------
// Flash attention. q,k: [B,H,T,HD] bf16. Q is pre-scaled by
// D^-.5 * HD^-.5 * log2(e)  so scores are already in the exp2 domain.
// k scaled by D^-.5. vt: [B,H,HD,T] bf16. out: [B,T,D] bf16.
// 128 threads = 4 waves; each wave owns a 16-row Q tile; 64 keys per step
// (16 WMMAs per softmax round).
// ---------------------------------------------------------------------------
__global__ __launch_bounds__(128) void flash_attn(
    const __bf16* __restrict__ q, const __bf16* __restrict__ k,
    const __bf16* __restrict__ vt, __bf16* __restrict__ out) {
  __shared__ __bf16 plds[4][16][64];  // per-wave P staging (D-frag -> A-frag)

  const int lane = threadIdx.x & 31;
  const int wid  = threadIdx.x >> 5;
  const int tiles_per_bh = cT / 64;
  const int bh    = blockIdx.x / tiles_per_bh;      // b*H + h
  const int qbase = (blockIdx.x % tiles_per_bh) * 64 + wid * 16;

  const __bf16* qp = q + ((size_t)bh * cT + qbase) * cHD;
  v16bf qf0 = load_a_bf16(qp, cHD, 0, 0,  lane);    // HD 0..31
  v16bf qf1 = load_a_bf16(qp, cHD, 0, 32, lane);    // HD 32..63

  float mrun[8], lrun[8], alph[8];
#pragma unroll
  for (int r = 0; r < 8; ++r) { mrun[r] = -INFINITY; lrun[r] = 0.0f; }
  v8f o[4];
#pragma unroll
  for (int nt = 0; nt < 4; ++nt) o[nt] = v8f_zero();

  const int rowh = (lane >> 4) << 3;                // +8 for upper lane half
  const int ncol = lane & 15;

  for (int kb = 0; kb < cT / 64; ++kb) {
    const int keybase = kb * 64;
    const __bf16* kp = k + ((size_t)bh * cT + keybase) * cHD;
    if (kb + 1 < cT / 64) {  // prefetch next K/V tiles
      __builtin_prefetch(kp + (size_t)64 * cHD + lane * 64, 0, 0);
      __builtin_prefetch(
          vt + ((size_t)bh * cHD + (lane & 15)) * cT + keybase + 64, 0, 0);
    }

    // S = Q * K^T  for 64 keys -> four 16x16 D-frags (8 WMMAs)
    v8f s[4];
#pragma unroll
    for (int nt = 0; nt < 4; ++nt) {
      v8f t = v8f_zero();
      t = wmma_bf16(qf0, load_b_rowK(kp, cHD, nt * 16, 0,  lane), t);
      t = wmma_bf16(qf1, load_b_rowK(kp, cHD, nt * 16, 32, lane), t);
      s[nt] = t;
    }

    // Online softmax per row (row = r + rowh; 16-lane groups share stats).
    // Scores are in the log2 domain (Q pre-scaled by log2e) -> plain exp2.
#pragma unroll
    for (int r = 0; r < 8; ++r) {
      float mx = fmaxf(fmaxf(s[0][r], s[1][r]), fmaxf(s[2][r], s[3][r]));
      mx = grp16_max(mx);
      const float mnew = fmaxf(mrun[r], mx);
      alph[r] = __builtin_exp2f(mrun[r] - mnew);
      const float p0 = __builtin_exp2f(s[0][r] - mnew);
      const float p1 = __builtin_exp2f(s[1][r] - mnew);
      const float p2 = __builtin_exp2f(s[2][r] - mnew);
      const float p3 = __builtin_exp2f(s[3][r] - mnew);
      float rs = grp16_sum((p0 + p1) + (p2 + p3));
      lrun[r] = lrun[r] * alph[r] + rs;
      mrun[r] = mnew;
      // stage P (bf16) into LDS, row-major [16][64]
      const int row = r + rowh;
      plds[wid][row][ncol]      = (__bf16)p0;
      plds[wid][row][ncol + 16] = (__bf16)p1;
      plds[wid][row][ncol + 32] = (__bf16)p2;
      plds[wid][row][ncol + 48] = (__bf16)p3;
    }

    // rescale running output
#pragma unroll
    for (int nt = 0; nt < 4; ++nt)
#pragma unroll
      for (int r = 0; r < 8; ++r) o[nt][r] *= alph[r];

    asm volatile("s_wait_dscnt 0" ::: "memory");

    // reload P as A-fragments (16x32 each, K halves of the 64-key block)
    v16bf pf0 = load_a_bf16(&plds[wid][0][0], 64, 0, 0,  lane);
    v16bf pf1 = load_a_bf16(&plds[wid][0][0], 64, 0, 32, lane);

    // O += P * V   (V^T stored [HD,T] -> contiguous B-frag loads)  (8 WMMAs)
#pragma unroll
    for (int nt = 0; nt < 4; ++nt) {
      const __bf16* vp = vt + ((size_t)bh * cHD + nt * 16 + ncol) * cT +
                         keybase + ((lane >> 4) << 4);
      v16bf vf0 = *(const v16bf*)vp;          // k = keybase+0 .. +31
      v16bf vf1 = *(const v16bf*)(vp + 32);   // k = keybase+32 .. +63
      o[nt] = wmma_bf16(pf0, vf0, o[nt]);
      o[nt] = wmma_bf16(pf1, vf1, o[nt]);
    }
  }

  // normalize (v_rcp_f32) and write out[b, t, h*HD + n]
  const int b = bh / cH;
  const int hcolbase = (bh % cH) * cHD;
  float inv[8];
#pragma unroll
  for (int r = 0; r < 8; ++r) inv[r] = __builtin_amdgcn_rcpf(lrun[r]);
#pragma unroll
  for (int nt = 0; nt < 4; ++nt) {
#pragma unroll
    for (int r = 0; r < 8; ++r) {
      const int t = qbase + r + rowh;
      const int n = nt * 16 + ncol;
      const float val = o[nt][r] * inv[r];
      out[((size_t)b * cT + t) * cD + hcolbase + n] = (__bf16)val;
    }
  }
}

// ---------------------------------------------------------------------------
// Readout: logits[b,v] = D^-.5 * sum_d h[b,T-1,d] * readout[v,d]   (fp32 out)
// ---------------------------------------------------------------------------
__global__ __launch_bounds__(256) void readout_kernel(
    const __bf16* __restrict__ h, const float* __restrict__ ro,
    float* __restrict__ out, float scale) {
  const int v = blockIdx.x * blockDim.x + threadIdx.x;
  const int b = blockIdx.y;
  const __bf16* hp = h + ((size_t)b * cT + (cT - 1)) * cD;
  const float* rp = ro + (size_t)v * cD;
  float s = 0.0f;
#pragma unroll 4
  for (int d = 0; d < cD; ++d) s += (float)hp[d] * rp[d];
  out[(size_t)b * cV + v] = s * scale;
}

// ---------------------------------------------------------------------------
// Launcher
// ---------------------------------------------------------------------------
extern "C" void kernel_launch(void* const* d_in, const int* in_sizes, int n_in,
                              void* d_out, int out_size, void* d_ws,
                              size_t ws_size, hipStream_t stream) {
  const float* x_f   = (const float*)d_in[0];   // [B,T,V]
  const float* te_f  = (const float*)d_in[1];   // [D,V]
  const float* pos_f = (const float*)d_in[2];   // [T,D]
  const float* wk_f  = (const float*)d_in[3];   // [L,D,D]
  const float* wq_f  = (const float*)d_in[4];
  const float* wv_f  = (const float*)d_in[5];
  const float* wp_f  = (const float*)d_in[6];
  const float* ro_f  = (const float*)d_in[7];   // [V,D]
  float* logits = (float*)d_out;                // [B,V]

  const size_t n_te = (size_t)cD * cV;
  const size_t n_w  = (size_t)cL * cD * cD;
  const size_t n_h  = (size_t)cB * cT * cD;

  __bf16* te_bf = (__bf16*)d_ws;
  __bf16* wk_bf = te_bf + n_te;
  __bf16* wq_bf = wk_bf + n_w;
  __bf16* wv_bf = wq_bf + n_w;
  __bf16* wp_bf = wv_bf + n_w;
  __bf16* hA    = wp_bf + n_w;
  __bf16* hBp   = hA + n_h;
  __bf16* qb    = hBp + n_h;
  __bf16* kb    = qb + n_h;
  __bf16* vtb   = kb + n_h;
  __bf16* att   = vtb + n_h;

  auto cvt = [&](const float* src, __bf16* dst, size_t n) {
    cvt_f32_bf16<<<(unsigned)((n + 255) / 256), 256, 0, stream>>>(src, dst, n);
  };
  cvt(te_f, te_bf, n_te);
  cvt(wk_f, wk_bf, n_w);
  cvt(wq_f, wq_bf, n_w);
  cvt(wv_f, wv_bf, n_w);
  cvt(wp_f, wp_bf, n_w);

  const int M = cB * cT;                       // 8192
  const float inv_sqrt_v  = 1.0f / sqrtf((float)cV);
  const float inv_sqrt_d  = 1.0f / sqrtf((float)cD);
  const float inv_sqrt_hd = 1.0f / sqrtf((float)cHD);
  const float log2e       = 1.44269504088896340736f;

  const dim3 gp(cD / 256, M / 64);             // (2, 128)

  // h0 = x @ te^T * V^-.5 + pos       (M=8192, N=512, K=2048)
  wmma_gemm<true, 0, true><<<gp, 256, 0, stream>>>(
      (const void*)x_f, te_bf, hA, M, cD, cV, inv_sqrt_v, pos_f);

  const __bf16* h = hA;
  __bf16* hn = hBp;
  for (int l = 0; l < cL; ++l) {
    const __bf16* Wq = wq_bf + (size_t)l * cD * cD;
    const __bf16* Wk = wk_bf + (size_t)l * cD * cD;
    const __bf16* Wv = wv_bf + (size_t)l * cD * cD;
    const __bf16* Wp = wp_bf + (size_t)l * cD * cD;

    // Q gets the HD^-.5 score scale and log2e (exp2-domain softmax) folded in.
    wmma_gemm<false, 1, false><<<gp, 256, 0, stream>>>(
        (const void*)h, Wq, qb, M, cD, cD, inv_sqrt_d * inv_sqrt_hd * log2e,
        nullptr);
    wmma_gemm<false, 1, false><<<gp, 256, 0, stream>>>(
        (const void*)h, Wk, kb, M, cD, cD, inv_sqrt_d, nullptr);
    wmma_gemm<false, 2, false><<<gp, 256, 0, stream>>>(
        (const void*)h, Wv, vtb, M, cD, cD, inv_sqrt_d, nullptr);

    flash_attn<<<cB * cH * (cT / 64), 128, 0, stream>>>(qb, kb, vtb, att);

    wmma_gemm<false, 0, false><<<gp, 256, 0, stream>>>(
        (const void*)att, Wp, hn, M, cD, cD, inv_sqrt_d, nullptr);

    const __bf16* tmp = h;
    h = hn;
    hn = (__bf16*)tmp;
  }

  readout_kernel<<<dim3(cV / 256, cB), 256, 0, stream>>>(h, ro_f, logits,
                                                         inv_sqrt_d);
}
